// Kernel_27797028339952
// MI455X (gfx1250) — compile-verified, matrix operand order checked by reading, not run on hardware
//
#include <hip/hip_runtime.h>

// CDNA5 / gfx1250 conv 5x5x3 (1 out channel) via banded-Toeplitz WMMA.
// D = A x B + C with A = f16 input rows (Toeplitz), B = banded f16 weights,
// C/D = f32 accumulator. 15 chained v_wmma_f32_16x16x32_f16 per 16x16 tile.

typedef __attribute__((ext_vector_type(16))) _Float16 v16h;
typedef __attribute__((ext_vector_type(8)))  _Float16 h8;
typedef __attribute__((ext_vector_type(4)))  _Float16 h4;
typedef __attribute__((ext_vector_type(8)))  float    v8f;

#define IMG   224
#define CH    3
#define KFW   5
#define TILE  32          // block output tile (2x2 wave quadrants of 16x16)
#define TPB   7           // tiles per image side: 224/32
#define SROWS 36          // staged rows per channel (32 + 2*2 halo)
#define SCOLS 36          // staged (valid) cols per channel
#define QCOLS 9           // SCOLS/4 for vectorized staging
#define SSTR  56          // LDS row stride in f16 (112B: 16B aligned, bank-friendly)

__global__ __launch_bounds__(128) void conv5_wmma(
    const float* __restrict__ x, const float* __restrict__ wlin,
    const float* __restrict__ bias, float* __restrict__ out, int ntiles)
{
  alignas(16) __shared__ _Float16 smem[CH * SROWS * SSTR];

  const int tid  = threadIdx.x;
  const int lane = tid & 31;
  const int wv   = tid >> 5;          // wave 0..3
  const int qy   = (wv >> 1) << 4;    // quadrant row offset (0/16)
  const int qx   = (wv & 1)  << 4;    // quadrant col offset (0/16)
  const int mn   = lane & 15;         // A: row m, B: col n, D: col n
  const int hi   = lane >> 4;         // K-half selector per WMMA VGPR layout

  // ---- Build the 15 banded B fragments once per wave (loop-invariant). ----
  // Element e of a v16h A/B fragment holds K = hi*8 + (e<8 ? e : e+8).
  // B[k][n] = w[c][kh][k-n] if 0 <= k-n <= 4 else 0  (rows k>=20 all zero).
  v16h bfr[CH * KFW];
#pragma unroll
  for (int c = 0; c < CH; ++c) {
#pragma unroll
    for (int kh = 0; kh < KFW; ++kh) {
      const int p = c * KFW + kh;
      const _Float16 w0 = (_Float16)wlin[p * KFW + 0];
      const _Float16 w1 = (_Float16)wlin[p * KFW + 1];
      const _Float16 w2 = (_Float16)wlin[p * KFW + 2];
      const _Float16 w3 = (_Float16)wlin[p * KFW + 3];
      const _Float16 w4 = (_Float16)wlin[p * KFW + 4];
      v16h bb = {};
#pragma unroll
      for (int e = 0; e < 16; ++e) {
        const int k = hi * 8 + (e < 8 ? e : e + 8);
        const int d = k - mn;
        _Float16 v = (_Float16)0.0f;
        v = (d == 0) ? w0 : v;
        v = (d == 1) ? w1 : v;
        v = (d == 2) ? w2 : v;
        v = (d == 3) ? w3 : v;
        v = (d == 4) ? w4 : v;
        bb[e] = v;
      }
      bfr[p] = bb;
    }
  }
  const float bv = bias[0];

  // ---- Zero LDS once: cols >= SCOLS are read as dead A data (B rows zero),
  //      must be finite (0) so 0*garbage never makes NaN. ----
  {
    const h8 z = {};
    for (int i = tid; i < (CH * SROWS * SSTR) / 8; i += 128)
      *(h8*)(&smem[i * 8]) = z;
  }

  // ---- Persistent tile loop: tiles = img * 49 + ty7 * 7 + tx7 ----
  for (int t = blockIdx.x; t < ntiles; t += gridDim.x) {
    const int img = t / (TPB * TPB);
    const int rem = t % (TPB * TPB);
    const int ty  = (rem / TPB) * TILE;
    const int tx  = (rem % TPB) * TILE;
    const float* __restrict__ xin = x + (size_t)img * (CH * IMG * IMG);
    __builtin_prefetch(xin + (size_t)(ty > 2 ? ty - 2 : 0) * IMG + tx, 0, 0);

    __syncthreads();  // previous tile's readers done before restaging

    // Stage 3ch x 36x36 halo (zero-padded at image edges) as f16 into LDS.
    for (int i = tid; i < CH * SROWS * QCOLS; i += 128) {
      const int c  = i / (SROWS * QCOLS);
      const int rr = (i / QCOLS) % SROWS;
      const int q  = i % QCOLS;
      const int gy  = ty + rr - 2;
      const int gxb = tx + 4 * q - 2;
      const bool rowok = (unsigned)gy < IMG;
      const float* src = xin + ((size_t)c * IMG + (rowok ? gy : 0)) * IMG;
      h4 pk;
#pragma unroll
      for (int j = 0; j < 4; ++j) {
        const int gx = gxb + j;
        const float v = (rowok && (unsigned)gx < IMG) ? src[gx] : 0.0f;
        pk[j] = (_Float16)v;
      }
      *(h4*)(&smem[(c * SROWS + rr) * SSTR + 4 * q]) = pk;
    }
    __syncthreads();

    // 15 chained WMMAs accumulate the full 5x5x3 conv for this 16x16 tile.
    v8f acc = {};
#pragma unroll
    for (int p = 0; p < CH * KFW; ++p) {
      const int c  = p / KFW;
      const int kh = p % KFW;
      // A[m][k] = smem[c][qy+m+kh][qx+k]; chunks are 16B-aligned ds_load_b128.
      const int off = (c * SROWS + (qy + mn + kh)) * SSTR + qx + hi * 8;
      const h8 a0 = *(const h8*)(&smem[off]);        // K = hi*8 + 0..7
      const h8 a1 = *(const h8*)(&smem[off + 16]);   // K = 16 + hi*8 + 0..7
      const v16h af = __builtin_shufflevector(a0, a1,
          0, 1, 2, 3, 4, 5, 6, 7, 8, 9, 10, 11, 12, 13, 14, 15);
      acc = __builtin_amdgcn_wmma_f32_16x16x32_f16(
          false, af, false, bfr[p], (short)0, acc, false, false);
    }

    // D layout: lane n = mn, VGPR r -> M = r + 8*hi.
    float* __restrict__ op =
        out + ((size_t)img * IMG + (ty + qy)) * IMG + tx + qx + mn;
#pragma unroll
    for (int r = 0; r < 8; ++r)
      op[(size_t)(r + hi * 8) * IMG] = acc[r] + bv;
  }
}

extern "C" void kernel_launch(void* const* d_in, const int* in_sizes, int n_in,
                              void* d_out, int out_size, void* d_ws, size_t ws_size,
                              hipStream_t stream) {
  (void)n_in; (void)out_size; (void)d_ws; (void)ws_size;
  const float* x    = (const float*)d_in[0];
  const float* wlin = (const float*)d_in[1];
  const float* b    = (const float*)d_in[2];
  float* out        = (float*)d_out;

  const int nimg   = in_sizes[0] / (CH * IMG * IMG);   // 256
  const int ntiles = nimg * TPB * TPB;                 // 12544 block tiles
  int nblk = ntiles / 8;                               // persistent: ~8 tiles/block
  if (nblk < 1) nblk = 1;
  if (nblk > ntiles) nblk = ntiles;

  conv5_wmma<<<dim3(nblk), dim3(128), 0, stream>>>(x, wlin, b, out, ntiles);
}
